// mAttention_52338471469579
// MI455X (gfx1250) — matrix operand order repaired and checked by reading, not compiled, not run on hardware
//
#include <hip/hip_runtime.h>
#include <cstdint>

// ---------------------------------------------------------------------------
// CDNA5 (gfx1250) gated attention, bf16 WMMA pipeline.
// Shapes: B=4 Q=1024 K=1024 C=512 H=8 HD=64 OUT=512.
// ---------------------------------------------------------------------------

typedef __attribute__((ext_vector_type(16))) __bf16 bf16x16;
typedef __attribute__((ext_vector_type(8)))  float  v8f;

#if defined(__AMDGCN__) && __has_builtin(__builtin_amdgcn_global_load_async_to_lds_b128)
#define USE_ASYNC_LDS 1
// exact parameter types per hipcc diagnostic: AS1 / AS3 pointers to <4 x i32>
typedef int v4i_gcc __attribute__((vector_size(16)));
typedef __attribute__((address_space(1))) v4i_gcc* async_gptr;
typedef __attribute__((address_space(3))) v4i_gcc* async_lptr;
#else
#define USE_ASYNC_LDS 0
#endif

union FragU { uint4 u[2]; bf16x16 v; };

__device__ __forceinline__ uint16_t f2bf(float f) {
  uint32_t u = __builtin_bit_cast(uint32_t, f);
  uint32_t r = u + 0x7FFFu + ((u >> 16) & 1u);   // round-to-nearest-even
  return (uint16_t)(r >> 16);
}

// pack two f32 -> two bf16 in one dword (low = a, high = b)
__device__ __forceinline__ uint32_t f2bf2(float a, float b) {
  return (uint32_t)f2bf(a) | ((uint32_t)f2bf(b) << 16);
}

__device__ __forceinline__ bf16x16 ldfrag(const uint16_t* p0, const uint16_t* p1) {
  FragU f;
  f.u[0] = *(const uint4*)p0;   // elements 0..7
  f.u[1] = *(const uint4*)p1;   // elements 8..15
  return f.v;
}

__device__ __forceinline__ v8f wmma_bf16(bf16x16 a, bf16x16 b, v8f c) {
  // D(16x16 f32) = A(16x32 bf16) x B(32x16 bf16) + C
  return __builtin_amdgcn_wmma_f32_16x16x32_bf16(
      /*neg_a=*/false, a, /*neg_b=*/false, b,
      /*c_mod=*/(short)0, c, /*reuse_a=*/false, /*reuse_b=*/false);
}

// ---------------------------------------------------------------------------
// Projection GEMM: C[4096x512] = A[4096x512] (f32->bf16) x W[512x512] (f32->bf16)
// mode 0: q  -> scale by 0.125, bf16 [b,h,row,hd]
// mode 1: k  -> bf16 [b,h,row,hd]
// mode 2: v  -> bf16 transposed [b,h,hd,key]
// mode 3: gate -> sigmoid(c + gating_b[n]) f32 [row, n]
// ---------------------------------------------------------------------------
__global__ __launch_bounds__(256)
void proj_gemm_kernel(const float* __restrict__ A, const float* __restrict__ W,
                      const float* __restrict__ gbias, void* __restrict__ dst,
                      int mode, float scale)
{
  __shared__ uint16_t lA[128][32];   // [row][k] bf16, 8 KiB
  __shared__ uint16_t lB[64][32];    // [n][k]  bf16 (transposed), 4 KiB

  const int tid  = threadIdx.x;
  const int m0   = blockIdx.x * 128;
  const int n0   = blockIdx.y * 64;
  const int wave = tid >> 5, lane = tid & 31;
  const int hig  = lane >> 4, l16 = lane & 15;

  v8f acc[4] = {{}, {}, {}, {}};

  for (int kk = 0; kk < 512; kk += 32) {
    // stage A tile 128x32 (f32 -> bf16), packed u32 LDS stores
    #pragma unroll
    for (int i = 0; i < 4; ++i) {
      int idx = tid + i * 256;
      int row = idx >> 3, c4 = (idx & 7) * 4;
      float4 av = *(const float4*)&A[(size_t)(m0 + row) * 512 + kk + c4];
      uint32_t* dp = (uint32_t*)&lA[row][c4];
      dp[0] = f2bf2(av.x, av.y);
      dp[1] = f2bf2(av.z, av.w);
    }
    // stage B tile 32x64 into LDS [n][k]: each thread takes two k-rows so the
    // k-pairs pack into single u32 stores (no d16_hi scatter).
    {
      int kr2 = tid >> 4;            // 0..15 -> k rows 2*kr2, 2*kr2+1
      int c4  = (tid & 15) * 4;
      const float* w0 = &W[(size_t)(kk + 2 * kr2) * 512 + n0 + c4];
      float4 wa4 = *(const float4*)w0;
      float4 wb4 = *(const float4*)(w0 + 512);
      *(uint32_t*)&lB[c4 + 0][2 * kr2] = f2bf2(wa4.x, wb4.x);
      *(uint32_t*)&lB[c4 + 1][2 * kr2] = f2bf2(wa4.y, wb4.y);
      *(uint32_t*)&lB[c4 + 2][2 * kr2] = f2bf2(wa4.z, wb4.z);
      *(uint32_t*)&lB[c4 + 3][2 * kr2] = f2bf2(wa4.w, wb4.w);
    }
    __syncthreads();

    // A frag: lanes 0-15 rows M=l16 with K {0..7,16..23}; lanes 16-31 K {8..15,24..31}
    const uint16_t* ar = &lA[wave * 16 + l16][hig * 8];
    bf16x16 a = ldfrag(ar, ar + 16);
    #pragma unroll
    for (int nt = 0; nt < 4; ++nt) {
      // B frag: lanes 0-15 K {0..7,16..23}, lanes 16-31 K {8..15,24..31}, N = l16
      const uint16_t* br = &lB[nt * 16 + l16][hig * 8];
      bf16x16 b = ldfrag(br, br + 16);
      acc[nt] = wmma_bf16(a, b, acc[nt]);
    }
    __syncthreads();
  }

  // epilogue: C layout -> element (M = i + 8*hig, N = l16) per 16x16 tile
  #pragma unroll
  for (int nt = 0; nt < 4; ++nt) {
    #pragma unroll
    for (int i = 0; i < 8; ++i) {
      int grow = m0 + wave * 16 + i + 8 * hig;
      int n    = n0 + nt * 16 + l16;
      int b    = grow >> 10, rq = grow & 1023;
      float c  = acc[nt][i];
      if (mode == 0) {
        uint16_t* q = (uint16_t*)dst;
        q[((size_t)(b * 8 + (n >> 6)) * 1024 + rq) * 64 + (n & 63)] = f2bf(c * scale);
      } else if (mode == 1) {
        uint16_t* kq = (uint16_t*)dst;
        kq[((size_t)(b * 8 + (n >> 6)) * 1024 + rq) * 64 + (n & 63)] = f2bf(c);
      } else if (mode == 2) {
        uint16_t* vq = (uint16_t*)dst;   // vT[b,h,hd,key]
        vq[((size_t)(b * 8 + (n >> 6)) * 64 + (n & 63)) * 1024 + rq] = f2bf(c);
      } else {
        float* g = (float*)dst;
        float x = c + gbias[n];
        g[(size_t)grow * 512 + n] = 1.0f / (1.0f + __expf(-x));
      }
    }
  }
}

// ---------------------------------------------------------------------------
// Attention: one wave per (b, h, 16-row query strip). Flash-style online
// softmax over 32-key chunks; writes raw S non-temporally to logits_update
// (write-once 128MB stream, keep L2 for nonbatched_bias); P bounced through
// LDS (C-layout -> A-layout) for the P*V WMMAs against vT.
// ---------------------------------------------------------------------------
__global__ __launch_bounds__(128)
void attn_kernel(const uint16_t* __restrict__ qb, const uint16_t* __restrict__ kb,
                 const uint16_t* __restrict__ vT, const float* __restrict__ gate,
                 const float* __restrict__ bias, const float* __restrict__ nb,
                 float* __restrict__ logits_out, uint16_t* __restrict__ wa_out)
{
  __shared__ uint16_t lP[4][16][32];   // per-wave P tile (16 rows x 32 keys) bf16

  const int tid  = threadIdx.x;
  const int wave = tid >> 5, lane = tid & 31;
  const int hig  = lane >> 4, l16 = lane & 15;
  const int wid  = blockIdx.x * 4 + wave;   // 0..2047
  const int b    = wid >> 9;
  const int h    = (wid >> 6) & 7;
  const int q0   = (wid & 63) * 16;

  const uint16_t* qbase = qb + ((size_t)(b * 8 + h) * 1024 + q0) * 64;
  const uint16_t* kbase = kb + (size_t)(b * 8 + h) * 1024 * 64;
  const uint16_t* vbase = vT + (size_t)(b * 8 + h) * 64 * 1024;
  const float*    nbb   = nb + ((size_t)h * 1024 + q0) * 1024;
  const float*    bb    = bias + (size_t)b * 1024;
  float*          lbase = logits_out + ((size_t)(b * 8 + h) * 1024 + q0) * 1024;

  // resident q A-frags covering channels 0..31 and 32..63
  bf16x16 aq0, aq1;
  {
    const uint16_t* r = qbase + (size_t)l16 * 64 + hig * 8;
    aq0 = ldfrag(r, r + 16);
    aq1 = ldfrag(r + 32, r + 48);
  }

  float m[8], ssum[8];
  #pragma unroll
  for (int i = 0; i < 8; ++i) { m[i] = -1e30f; ssum[i] = 0.0f; }
  v8f acc[4] = {{}, {}, {}, {}};

  for (int kt2 = 0; kt2 < 32; ++kt2) {
    // prefetch next chunk's k rows (32 rows via both lane halves) and v rows
    if (kt2 + 1 < 32) {
      __builtin_prefetch(kbase + (size_t)(kt2 * 32 + 32 + hig * 16 + l16) * 64, 0, 0);
      __builtin_prefetch(vbase + (size_t)lane * 1024 + kt2 * 32 + 32, 0, 0);
      __builtin_prefetch(vbase + (size_t)(lane + 32) * 1024 + kt2 * 32 + 32, 0, 0);
    }

    float L[2][8];
    #pragma unroll
    for (int half = 0; half < 2; ++half) {
      const int k0  = kt2 * 32 + half * 16;
      const int key = k0 + l16;
      // k B-frags: N = key, K = channel (contiguous in k row)
      const uint16_t* kr = kbase + (size_t)key * 64 + hig * 8;
      bf16x16 bk0 = ldfrag(kr, kr + 16);         // c 0..31
      bf16x16 bk1 = ldfrag(kr + 32, kr + 48);    // c 32..63
      v8f s = {};
      s = wmma_bf16(aq0, bk0, s);
      s = wmma_bf16(aq1, bk1, s);
      float bv = bb[k0 + l16];
      #pragma unroll
      for (int i = 0; i < 8; ++i) {
        int M = i + 8 * hig;
        float sv = s[i];
        __builtin_nontemporal_store(sv, &lbase[(size_t)M * 1024 + k0 + l16]);
        L[half][i] = sv + bv + nbb[(size_t)M * 1024 + k0 + l16]; // full logits
      }
    }

    // joint online softmax update over the 32 new columns
    #pragma unroll
    for (int i = 0; i < 8; ++i) {
      float mx = fmaxf(L[0][i], L[1][i]);
      #pragma unroll
      for (int off = 1; off < 16; off <<= 1)
        mx = fmaxf(mx, __shfl_xor(mx, off, 32));   // reduce within 16-lane group
      float mn   = fmaxf(m[i], mx);
      float corr = __expf(m[i] - mn);
      m[i] = mn;
      float pa = __expf(L[0][i] - mn);
      float pb = __expf(L[1][i] - mn);
      float rs = pa + pb;
      #pragma unroll
      for (int off = 1; off < 16; off <<= 1)
        rs += __shfl_xor(rs, off, 32);
      ssum[i] = ssum[i] * corr + rs;
      #pragma unroll
      for (int nt = 0; nt < 4; ++nt) acc[nt][i] *= corr;
      int M = i + 8 * hig;
      lP[wave][M][l16]      = f2bf(pa);
      lP[wave][M][16 + l16] = f2bf(pb);
    }
    asm volatile("s_wait_dscnt 0x0" ::: "memory");  // LDS stores visible to frag reads

    // P as A-frag (16x32 over the 32 keys of this chunk)
    const uint16_t* pr = &lP[wave][l16][hig * 8];
    bf16x16 ap = ldfrag(pr, pr + 16);

    // V B-frags from vT[b,h,hd,key]: N = hd, K = key (contiguous in vT row)
    #pragma unroll
    for (int nt = 0; nt < 4; ++nt) {
      const uint16_t* vr = vbase + (size_t)(nt * 16 + l16) * 1024 + kt2 * 32 + hig * 8;
      bf16x16 bv = ldfrag(vr, vr + 16);
      acc[nt] = wmma_bf16(ap, bv, acc[nt]);
    }
  }

  // epilogue: normalize, gate, store bf16 for the output GEMM
  #pragma unroll
  for (int nt = 0; nt < 4; ++nt) {
    #pragma unroll
    for (int i = 0; i < 8; ++i) {
      int M   = i + 8 * hig;
      int col = nt * 16 + l16;
      float wa = acc[nt][i] / ssum[i];
      size_t gidx = ((size_t)(b * 1024 + q0 + M) * 512) + h * 64 + col;
      wa_out[gidx] = f2bf(wa * gate[gidx]);
    }
  }
}

// ---------------------------------------------------------------------------
// Output GEMM: out[4096x512] = wa_bf16[4096x512] x output_w(bf16) + output_b
// A tile is already bf16 -> stage it with async global->LDS DMA when available.
// ---------------------------------------------------------------------------
__global__ __launch_bounds__(256)
void out_gemm_kernel(const uint16_t* __restrict__ Ab, const float* __restrict__ W,
                     const float* __restrict__ obias, float* __restrict__ out)
{
  __shared__ uint16_t lA[128][32];
  __shared__ uint16_t lB[64][32];

  const int tid  = threadIdx.x;
  const int m0   = blockIdx.x * 128;
  const int n0   = blockIdx.y * 64;
  const int wave = tid >> 5, lane = tid & 31;
  const int hig  = lane >> 4, l16 = lane & 15;

  v8f acc[4] = {{}, {}, {}, {}};

  for (int kk = 0; kk < 512; kk += 32) {
    // stage A (bf16 copy): async DMA straight into LDS if the builtin exists
    #pragma unroll
    for (int i = 0; i < 2; ++i) {
      int idx = tid + i * 256;
      int row = idx >> 2, q4 = (idx & 3) * 8;
      const uint16_t* gsrc = &Ab[(size_t)(m0 + row) * 512 + kk + q4];
#if USE_ASYNC_LDS
      __builtin_amdgcn_global_load_async_to_lds_b128(
          (async_gptr)(void*)gsrc,
          (async_lptr)(void*)&lA[row][q4],
          0, 0);
#else
      *(uint4*)&lA[row][q4] = *(const uint4*)gsrc;
#endif
    }
    // stage B (f32 -> bf16, transposed [n][k], packed u32 stores)
    {
      int kr2 = tid >> 4;
      int c4  = (tid & 15) * 4;
      const float* w0 = &W[(size_t)(kk + 2 * kr2) * 512 + n0 + c4];
      float4 wa4 = *(const float4*)w0;
      float4 wb4 = *(const float4*)(w0 + 512);
      *(uint32_t*)&lB[c4 + 0][2 * kr2] = f2bf2(wa4.x, wb4.x);
      *(uint32_t*)&lB[c4 + 1][2 * kr2] = f2bf2(wa4.y, wb4.y);
      *(uint32_t*)&lB[c4 + 2][2 * kr2] = f2bf2(wa4.z, wb4.z);
      *(uint32_t*)&lB[c4 + 3][2 * kr2] = f2bf2(wa4.w, wb4.w);
    }
#if USE_ASYNC_LDS
    asm volatile("s_wait_asynccnt 0x0" ::: "memory");
#endif
    __syncthreads();

    const uint16_t* ar = &lA[wave * 16 + l16][hig * 8];
    bf16x16 a = ldfrag(ar, ar + 16);
    #pragma unroll
    for (int nt = 0; nt < 4; ++nt) {
      const uint16_t* br = &lB[nt * 16 + l16][hig * 8];
      bf16x16 b = ldfrag(br, br + 16);
      acc[nt] = wmma_bf16(a, b, acc[nt]);
    }
    __syncthreads();
  }
  #pragma unroll
  for (int nt = 0; nt < 4; ++nt) {
    #pragma unroll
    for (int i = 0; i < 8; ++i) {
      int grow = m0 + wave * 16 + i + 8 * hig;
      int n    = n0 + nt * 16 + l16;
      out[(size_t)grow * 512 + n] = acc[nt][i] + obias[n];
    }
  }
}

// ---------------------------------------------------------------------------
extern "C" void kernel_launch(void* const* d_in, const int* in_sizes, int n_in,
                              void* d_out, int out_size, void* d_ws, size_t ws_size,
                              hipStream_t stream) {
  const float* q_data   = (const float*)d_in[0];
  const float* m_data   = (const float*)d_in[1];
  const float* bias     = (const float*)d_in[2];
  const float* nbias    = (const float*)d_in[3];
  const float* query_w  = (const float*)d_in[4];
  const float* key_w    = (const float*)d_in[5];
  const float* value_w  = (const float*)d_in[6];
  const float* gating_w = (const float*)d_in[7];
  const float* gating_b = (const float*)d_in[8];
  const float* output_w = (const float*)d_in[9];
  const float* output_b = (const float*)d_in[10];

  float* out_main   = (float*)d_out;                       // [4,1024,512]
  float* logits_out = out_main + (size_t)4 * 1024 * 512;   // [4,8,1024,1024]

  char* ws = (char*)d_ws;
  uint16_t* qb   = (uint16_t*)(ws);                  // 4 MiB bf16 [b,h,q,hd] (scaled)
  uint16_t* kb   = (uint16_t*)(ws + (4u  << 20));    // 4 MiB bf16 [b,h,k,hd]
  uint16_t* vT   = (uint16_t*)(ws + (8u  << 20));    // 4 MiB bf16 [b,h,hd,k]
  float*    gate = (float*)   (ws + (12u << 20));    // 8 MiB f32  [b*q, h*hd]
  uint16_t* wa   = (uint16_t*)(ws + (20u << 20));    // 4 MiB bf16 [b*q, h*hd]

  dim3 gridG(32, 8), blockG(256);
  hipLaunchKernelGGL(proj_gemm_kernel, gridG, blockG, 0, stream,
                     q_data, query_w, (const float*)nullptr, (void*)qb, 0, 0.125f);
  hipLaunchKernelGGL(proj_gemm_kernel, gridG, blockG, 0, stream,
                     m_data, key_w,   (const float*)nullptr, (void*)kb, 1, 1.0f);
  hipLaunchKernelGGL(proj_gemm_kernel, gridG, blockG, 0, stream,
                     m_data, value_w, (const float*)nullptr, (void*)vT, 2, 1.0f);
  hipLaunchKernelGGL(proj_gemm_kernel, gridG, blockG, 0, stream,
                     q_data, gating_w, gating_b, (void*)gate, 3, 1.0f);

  hipLaunchKernelGGL(attn_kernel, dim3(512), dim3(128), 0, stream,
                     qb, kb, vT, gate, bias, nbias, logits_out, wa);

  hipLaunchKernelGGL(out_gemm_kernel, gridG, blockG, 0, stream,
                     wa, output_w, output_b, out_main);
}
